// GNNNodeEmbedding_2362232012848
// MI455X (gfx1250) — compile-verified, matrix operand order checked by reading, not run on hardware
//
#include <hip/hip_runtime.h>
#include <hip/hip_bf16.h>

typedef __attribute__((ext_vector_type(16))) _Float16 v16h;
typedef __attribute__((ext_vector_type(2)))  _Float16 h2;
typedef __attribute__((ext_vector_type(8)))  float    v8f;

#define N_NODES 50000
#define N_PAD   50048   /* 391 * 128 */
#define N_EDGES 800000
#define D 64
#define L_LAYERS 5
#define XDIM 92
#define EADIM 50
#define BN_EPS 1e-5f
#define ASTR 66   /* LDS row stride (halfs) for K<=64 tiles: 33 dwords, odd */
#define XSTR 98   /* LDS row stride (halfs) for K=96 tiles: 49 dwords, odd */

static_assert(N_EDGES % 128 == 0, "edge kernel assumes full 128-edge blocks");
static_assert(N_PAD % 128 == 0, "node blocks must be full in padded space");

// ---- WMMA fragment loader (wave32, 16x16x32 f16) -------------------------
// A (16x32): lanes 0-15 rows M=0..15; lanes 16-31 same rows, K+8.
// VGPR pair r: r<4 -> K=2r ; r>=4 -> K=16+2(r-4). B mirrored via transposed
// weight staging, so both A and B fragments are 8 contiguous 32-bit LDS loads.
__device__ __forceinline__ v16h load_frag(const _Float16* __restrict__ base) {
  const int half = (threadIdx.x >> 4) & 1;
  union { v16h v; h2 p[8]; } u;
#pragma unroll
  for (int r = 0; r < 8; ++r) {
    int kp = ((r < 4) ? 2 * r : 16 + 2 * (r - 4)) + 8 * half;
    u.p[r] = *(const h2*)(base + kp);
  }
  return u.v;
}

__device__ __forceinline__ v8f wmma16(v16h a, v16h b, v8f c) {
  return __builtin_amdgcn_wmma_f32_16x16x32_f16(false, a, false, b, (short)0, c,
                                                false, false);
}

// ---- Kernel 1: atom encoder  h = x @ W + b ; agg = (1+eps0)*h ------------
__global__ __launch_bounds__(256) void k_atom_encoder(
    const float* __restrict__ x, const float* __restrict__ atom_W,
    const float* __restrict__ atom_b, const float* __restrict__ eps,
    float* __restrict__ h, float* __restrict__ aggbuf,
    float* __restrict__ stats) {
  __shared__ _Float16 AS[128 * XSTR];   // 128 rows x 96 K (padded stride)
  __shared__ _Float16 Wt[D * XSTR];     // transposed: Wt[n][k]
  __shared__ float bs[D];
  const int tid = threadIdx.x;
  const int rowbase0 = blockIdx.x * 128;

  // stage transposed weights, K padded 92 -> 96
  for (int i = tid; i < 96 * D; i += 256) {
    int n = i & 63, k = i >> 6;  // consecutive tid -> consecutive n: coalesced
    Wt[n * XSTR + k] = (_Float16)((k < XDIM) ? atom_W[k * D + n] : 0.0f);
  }
  if (tid < D) bs[tid] = atom_b[tid];
  if (blockIdx.x == 0 && tid < 2 * D) stats[tid] = 0.0f;

  // stage x tile: flat float4 over the block's contiguous region
  const int nrows = min(128, N_NODES - rowbase0);
  const int flatlim = nrows * XDIM;          // multiple of 4
  const float* xb = x + (size_t)rowbase0 * XDIM;
  for (int j = tid * 4; j < flatlim; j += 1024) {
    float4 v = *(const float4*)(xb + j);
    float vv[4] = {v.x, v.y, v.z, v.w};
#pragma unroll
    for (int t = 0; t < 4; ++t) {
      int row = (j + t) / XDIM, col = (j + t) % XDIM;
      AS[row * XSTR + col] = (_Float16)vv[t];
    }
  }
  for (int i = tid; i < nrows * 4; i += 256) {  // zero K-pad cols 92..95
    int row = i >> 2, col = XDIM + (i & 3);
    AS[row * XSTR + col] = (_Float16)0.0f;
  }
  for (int i = tid; i < (128 - nrows) * 96; i += 256) {  // zero pad rows
    int row = nrows + i / 96, col = i % 96;
    AS[row * XSTR + col] = (_Float16)0.0f;
  }
  __syncthreads();

  const int lane = tid & 31, wave = tid >> 5;
  const int half = lane >> 4, l15 = lane & 15;
  const _Float16* arow = AS + (wave * 16 + l15) * XSTR;

  v8f c[4] = {};
#pragma unroll
  for (int ks = 0; ks < 3; ++ks) {
    v16h a = load_frag(arow + ks * 32);
#pragma unroll
    for (int nt = 0; nt < 4; ++nt) {
      v16h b = load_frag(Wt + (nt * 16 + l15) * XSTR + ks * 32);
      c[nt] = wmma16(a, b, c[nt]);
    }
  }
  const float e0 = 1.0f + eps[0];
  const int rowbase = rowbase0 + wave * 16;
#pragma unroll
  for (int nt = 0; nt < 4; ++nt) {
    int col = nt * 16 + l15;
#pragma unroll
    for (int r = 0; r < 8; ++r) {
      int row = rowbase + r + 8 * half;              // < N_PAD always
      float v = (row < N_NODES) ? (c[nt][r] + bs[col]) : 0.0f;
      size_t off = (size_t)row * D + col;
      h[off] = v;
      aggbuf[off] = e0 * v;
    }
  }
}

// ---- Kernel 2: fused edge GEMM + gather + relu + scatter-add -------------
__global__ __launch_bounds__(256) void k_edge_message(
    const float* __restrict__ edge_attr, const long long* __restrict__ edge_index,
    const float* __restrict__ edge_W, const float* __restrict__ edge_b,
    const float* __restrict__ h, float* __restrict__ agg) {
  __shared__ _Float16 AS[128 * ASTR];
  __shared__ _Float16 Wt[D * ASTR];
  __shared__ float bs[D];
  __shared__ int srcs[128], dsts[128];
  const int tid = threadIdx.x;
  const int eblock = blockIdx.x * 128;
  const float* eap = edge_attr + (size_t)eblock * EADIM;

  for (int i = tid; i < D * D; i += 256) {
    int n = i & 63, k = i >> 6;
    Wt[n * ASTR + k] = (_Float16)((k < EADIM) ? edge_W[k * D + n] : 0.0f);
  }
  if (tid < D) bs[tid] = edge_b[tid];
  if (tid < 128) {
    srcs[tid] = (int)edge_index[eblock + tid];
    dsts[tid] = (int)edge_index[N_EDGES + eblock + tid];
  }
  // stage edge_attr tile: 6400 contiguous floats, float4 coalesced
  for (int j = tid * 4; j < 128 * EADIM; j += 1024) {
    float4 v = *(const float4*)(eap + j);
    float vv[4] = {v.x, v.y, v.z, v.w};
#pragma unroll
    for (int t = 0; t < 4; ++t) {
      int row = (j + t) / EADIM, col = (j + t) % EADIM;
      AS[row * ASTR + col] = (_Float16)vv[t];
    }
  }
  for (int i = tid; i < 128 * 14; i += 256) {  // zero K-pad cols 50..63
    int row = i / 14, col = EADIM + i % 14;
    AS[row * ASTR + col] = (_Float16)0.0f;
  }
  if (eblock + 256 <= N_EDGES)  // gfx1250 global_prefetch of next tile
    __builtin_prefetch(eap + 128 * EADIM + tid * 25, 0, 1);
  __syncthreads();

  const int lane = tid & 31, wave = tid >> 5;
  const int half = lane >> 4, l15 = lane & 15;
  const _Float16* arow = AS + (wave * 16 + l15) * ASTR;

  v8f c[4] = {};
#pragma unroll
  for (int ks = 0; ks < 2; ++ks) {
    v16h a = load_frag(arow + ks * 32);
#pragma unroll
    for (int nt = 0; nt < 4; ++nt) {
      v16h b = load_frag(Wt + (nt * 16 + l15) * ASTR + ks * 32);
      c[nt] = wmma16(a, b, c[nt]);
    }
  }
  // branch-free epilogue: bias + gather h[src] + relu + atomic scatter
#pragma unroll
  for (int nt = 0; nt < 4; ++nt) {
    int col = nt * 16 + l15;
#pragma unroll
    for (int r = 0; r < 8; ++r) {
      int em = wave * 16 + r + 8 * half;
      float msg =
          fmaxf(c[nt][r] + bs[col] + h[(size_t)srcs[em] * D + col], 0.0f);
      unsafeAtomicAdd(&agg[(size_t)dsts[em] * D + col], msg);
    }
  }
}

// ---- Kernel 3: z = relu(zp@W1+b1)@W2+b2 ; block-reduced BN stats ---------
__global__ __launch_bounds__(256) void k_mlp_stats(
    const float* __restrict__ zp, const float* __restrict__ W1,
    const float* __restrict__ b1, const float* __restrict__ W2,
    const float* __restrict__ b2, float* __restrict__ zout,
    float* __restrict__ stats) {
  __shared__ _Float16 AS[128 * ASTR];  // zp tile, then reused as y stage
  __shared__ _Float16 W1t[D * ASTR], W2t[D * ASTR];
  __shared__ float b1s[D], b2s[D];
  __shared__ float st[2 * D];
  const int tid = threadIdx.x;
  const int rowbase0 = blockIdx.x * 128;

  for (int i = tid; i < D * D; i += 256) {
    int n = i & 63, k = i >> 6;
    W1t[n * ASTR + k] = (_Float16)W1[k * D + n];
    W2t[n * ASTR + k] = (_Float16)W2[k * D + n];
  }
  if (tid < D) { b1s[tid] = b1[tid]; b2s[tid] = b2[tid]; }
  if (tid < 2 * D) st[tid] = 0.0f;
  // stage zp tile (padded buffer: always 128 full rows readable)
  const float* zb = zp + (size_t)rowbase0 * D;
  for (int j = tid * 4; j < 128 * D; j += 1024) {
    float4 v = *(const float4*)(zb + j);
    int row = j >> 6, col = j & 63;
    h2* dst = (h2*)(AS + row * ASTR + col);
    h2 p0; p0.x = (_Float16)v.x; p0.y = (_Float16)v.y;
    h2 p1; p1.x = (_Float16)v.z; p1.y = (_Float16)v.w;
    dst[0] = p0;
    dst[1] = p1;
  }
  __syncthreads();

  const int lane = tid & 31, wave = tid >> 5;
  const int half = lane >> 4, l15 = lane & 15;
  const _Float16* arow = AS + (wave * 16 + l15) * ASTR;

  // GEMM 1: y = relu(zp @ W1 + b1)
  v8f c1[4] = {};
#pragma unroll
  for (int ks = 0; ks < 2; ++ks) {
    v16h a = load_frag(arow + ks * 32);
#pragma unroll
    for (int nt = 0; nt < 4; ++nt) {
      v16h b = load_frag(W1t + (nt * 16 + l15) * ASTR + ks * 32);
      c1[nt] = wmma16(a, b, c1[nt]);
    }
  }
  __syncthreads();  // all waves done reading zp tile before overwrite
#pragma unroll
  for (int nt = 0; nt < 4; ++nt) {
    int col = nt * 16 + l15;
#pragma unroll
    for (int r = 0; r < 8; ++r) {
      int m = r + 8 * half;
      AS[(wave * 16 + m) * ASTR + col] =
          (_Float16)fmaxf(c1[nt][r] + b1s[col], 0.0f);
    }
  }
  __syncthreads();  // C-layout -> A-layout transpose via LDS

  // GEMM 2: z = y @ W2 + b2
  v8f c2[4] = {};
#pragma unroll
  for (int ks = 0; ks < 2; ++ks) {
    v16h a = load_frag(arow + ks * 32);
#pragma unroll
    for (int nt = 0; nt < 4; ++nt) {
      v16h b = load_frag(W2t + (nt * 16 + l15) * ASTR + ks * 32);
      c2[nt] = wmma16(a, b, c2[nt]);
    }
  }
  const int rowbase = rowbase0 + wave * 16;
#pragma unroll
  for (int nt = 0; nt < 4; ++nt) {
    int col = nt * 16 + l15;
    float s = 0.0f, s2 = 0.0f;
#pragma unroll
    for (int r = 0; r < 8; ++r) {
      int row = rowbase + r + 8 * half;  // < N_PAD always
      float v = c2[nt][r] + b2s[col];
      zout[(size_t)row * D + col] = v;                   // padded, unguarded
      float w = (row < N_NODES) ? 1.0f : 0.0f;           // branchless mask
      s += v * w;
      s2 += v * v * w;
    }
    unsafeAtomicAdd(&st[col], s);      // LDS ds_add_f32
    unsafeAtomicAdd(&st[D + col], s2);
  }
  __syncthreads();
  if (tid < 2 * D) unsafeAtomicAdd(&stats[tid], st[tid]);
}

// ---- Kernel 4: fold BN stats into per-channel scale/shift ----------------
__global__ __launch_bounds__(64) void k_bn_finalize(
    const float* __restrict__ stats, const float* __restrict__ gamma,
    const float* __restrict__ beta, float* __restrict__ coef) {
  int t = threadIdx.x;
  if (t < D) {
    const float invN = 1.0f / (float)N_NODES;
    float mu = stats[t] * invN;
    float var = stats[D + t] * invN - mu * mu;
    float a = gamma[t] * rsqrtf(var + BN_EPS);
    coef[t] = a;
    coef[D + t] = beta[t] - mu * a;
  }
}

// ---- Kernel 5: BN (+relu), prep next agg, keep pads zero, zero stats -----
__global__ __launch_bounds__(256) void k_bn_apply(
    const float* __restrict__ zin, float* __restrict__ hout,
    const float* __restrict__ coef, float* __restrict__ aggnext,
    const float* __restrict__ eps_next, float* __restrict__ stats,
    int do_relu, int pad_out) {
  size_t idx = (size_t)blockIdx.x * 256 + threadIdx.x;
  if (idx < (size_t)N_NODES * D) {
    int col = (int)(idx & 63);
    float v = zin[idx] * coef[col] + coef[D + col];
    if (do_relu) v = fmaxf(v, 0.0f);
    hout[idx] = v;
    if (aggnext) aggnext[idx] = (1.0f + eps_next[0]) * v;
  } else if (idx < (size_t)N_PAD * D) {
    if (pad_out) hout[idx] = 0.0f;
    if (aggnext) aggnext[idx] = 0.0f;
  }
  if (blockIdx.x == 0 && threadIdx.x < 2 * D) stats[threadIdx.x] = 0.0f;
}

// ---- Host orchestration ---------------------------------------------------
extern "C" void kernel_launch(void* const* d_in, const int* in_sizes, int n_in,
                              void* d_out, int out_size, void* d_ws,
                              size_t ws_size, hipStream_t stream) {
  (void)in_sizes; (void)n_in; (void)out_size; (void)ws_size;
  const float* x         = (const float*)d_in[0];
  const long long* eidx  = (const long long*)d_in[1];
  const float* edge_attr = (const float*)d_in[2];
  const float* atom_W    = (const float*)d_in[3];
  const float* atom_b    = (const float*)d_in[4];
  const float* edge_W    = (const float*)d_in[5];
  const float* edge_b    = (const float*)d_in[6];
  const float* mlp_W1    = (const float*)d_in[7];
  const float* mlp_b1    = (const float*)d_in[8];
  const float* mlp_W2    = (const float*)d_in[9];
  const float* mlp_b2    = (const float*)d_in[10];
  const float* eps       = (const float*)d_in[11];
  const float* bn_gamma  = (const float*)d_in[12];
  const float* bn_beta   = (const float*)d_in[13];
  float* out = (float*)d_out;

  float* ws    = (float*)d_ws;
  float* buf0  = ws;                               // [N_PAD, D]
  float* buf1  = buf0 + (size_t)N_PAD * D;         // [N_PAD, D]
  float* agg   = buf1 + (size_t)N_PAD * D;         // [N_PAD, D]
  float* stats = agg + (size_t)N_PAD * D;          // [2*D]
  float* coef  = stats + 2 * D;                    // [2*D]

  dim3 blk(256);
  const int nodeBlocks = N_PAD / 128;                    // 391
  const int edgeBlocks = N_EDGES / 128;                  // 6250
  const int ewBlocks   = (int)(((size_t)N_PAD * D + 255) / 256);

  k_atom_encoder<<<nodeBlocks, blk, 0, stream>>>(x, atom_W, atom_b, eps, buf0,
                                                 agg, stats);

  float* hbuf[2] = {buf0, buf1};
  for (int l = 0; l < L_LAYERS; ++l) {
    float* hin  = hbuf[l & 1];
    float* zbuf = hbuf[(l + 1) & 1];
    int last    = (l == L_LAYERS - 1);
    float* hout = last ? out : zbuf;

    k_edge_message<<<edgeBlocks, blk, 0, stream>>>(
        edge_attr, eidx, edge_W + (size_t)l * EADIM * D,
        edge_b + (size_t)l * D, hin, agg);
    k_mlp_stats<<<nodeBlocks, blk, 0, stream>>>(
        agg, mlp_W1 + (size_t)l * D * D, mlp_b1 + (size_t)l * D,
        mlp_W2 + (size_t)l * D * D, mlp_b2 + (size_t)l * D, zbuf, stats);
    k_bn_finalize<<<1, 64, 0, stream>>>(stats, bn_gamma + (size_t)l * D,
                                        bn_beta + (size_t)l * D, coef);
    k_bn_apply<<<ewBlocks, blk, 0, stream>>>(zbuf, hout, coef,
                                             last ? nullptr : agg,
                                             eps + (last ? l : l + 1), stats,
                                             !last, !last);
  }
}
